// NTXentLoss_86268713108397
// MI455X (gfx1250) — matrix operand order repaired and checked by reading, not compiled, not run on hardware
//
#include <hip/hip_runtime.h>
#include <stdint.h>

// ---------------------------------------------------------------------------
// NT-Xent loss, fused on gfx1250 (MI455X).
//   K1: row-normalize z1,z2 (f32) -> zh (f16, 8192x256) in workspace
//   K2: 128 blocks x 8 waves. Wave (rt,par) owns row tile r_w and sweeps
//       column tiles of its parity. B tiles stream global->LDS with
//       global_load_async_to_lds_b128 (ASYNCcnt), double buffered; compute
//       stages 8 B fragments from LDS (sched_barrier-fenced) then issues 8
//       back-to-back v_wmma_f32_16x16x32_f16, fused mask/pos/fixed-shift LSE.
//   K3: deterministic tree reduction -> mean loss scalar
// ---------------------------------------------------------------------------

typedef __attribute__((ext_vector_type(16))) _Float16 v16h;
typedef __attribute__((ext_vector_type(8)))  _Float16 v8h;
typedef __attribute__((ext_vector_type(8)))  float    v8f;
typedef __attribute__((ext_vector_type(4)))  float    v4f;

#define NT_B     4096
#define NT_N     8192          // 2*B
#define NT_D     256
#define NT_NBLK  (NT_N / 16)   // 512 row/col tiles of 16
#define NT_BBLK  (NT_B / 16)   // 256 tile offset to the positive pair
#define NT_STEPS (NT_NBLK / 2) // 256 tile-pair steps per block
#define NT_INVT  2.0f          // 1 / temperature(0.5)
#define NT_SHIFT 2.0f          // |sim| <= 1  =>  |logit| <= 2 (fixed lse shift)

#define LDS_ROW   264          // 256 f16 + 8 f16 pad = 528 B (4 banks mod 64)
#define LDS_ROWB  (LDS_ROW * 2)
#define BUF_HALF  (16 * LDS_ROW)           // one 16-row tile, padded (elements)

// --- CDNA5 async global->LDS copy (ASYNCcnt), 16B per lane -----------------
__device__ __forceinline__ void async_b128(uint32_t lds_off, const void* gptr) {
  asm volatile("global_load_async_to_lds_b128 %0, %1, off"
               :: "v"(lds_off), "v"(gptr) : "memory");
}
template <int N>
__device__ __forceinline__ void wait_asynccnt() {
#if __has_builtin(__builtin_amdgcn_s_wait_asynccnt)
  __builtin_amdgcn_s_wait_asynccnt(N);
#else
  asm volatile("s_wait_asynccnt %0" :: "i"(N) : "memory");
#endif
}
__device__ __forceinline__ void sched_fence() {
#if __has_builtin(__builtin_amdgcn_sched_barrier)
  __builtin_amdgcn_sched_barrier(0);   // nothing may be moved across
#else
  asm volatile("" ::: "memory");
#endif
}

// ---------------- Kernel 1: normalize rows, convert to f16 ------------------
__global__ void nt_normalize(const float* __restrict__ z1,
                             const float* __restrict__ z2,
                             _Float16* __restrict__ zh) {
  const int row  = blockIdx.x * 8 + (threadIdx.x >> 5);
  const int lane = threadIdx.x & 31;
  const float* src = (row < NT_B) ? (z1 + (size_t)row * NT_D)
                                  : (z2 + (size_t)(row - NT_B) * NT_D);
  v4f a = *(const v4f*)(src + lane * 8);
  v4f b = *(const v4f*)(src + lane * 8 + 4);
  float ss = a.x*a.x + a.y*a.y + a.z*a.z + a.w*a.w
           + b.x*b.x + b.y*b.y + b.z*b.z + b.w*b.w;
#pragma unroll
  for (int m = 1; m < 32; m <<= 1) ss += __shfl_xor(ss, m, 32);
  const float rn = rsqrtf(ss);
  v8h o;
  o[0] = (_Float16)(a.x * rn); o[1] = (_Float16)(a.y * rn);
  o[2] = (_Float16)(a.z * rn); o[3] = (_Float16)(a.w * rn);
  o[4] = (_Float16)(b.x * rn); o[5] = (_Float16)(b.y * rn);
  o[6] = (_Float16)(b.z * rn); o[7] = (_Float16)(b.w * rn);
  *(v8h*)(zh + (size_t)row * NT_D + lane * 8) = o;
}

// ---------------- Kernel 2: fused WMMA gram + masked logsumexp --------------
// 128 blocks x 256 threads. Wave w: row tile r_w = bid*4 + (w>>1),
// column parity par = w&1 (256 column tiles each). B tile-pairs (32 rows,
// 16 KB) are async-copied to double-buffered LDS and shared by all 8 waves.
__global__ __launch_bounds__(256) void nt_lse(const _Float16* __restrict__ zh,
                                              float* __restrict__ row_loss) {
  __shared__ _Float16 sB[2][2][16][LDS_ROW];   // [buf][parity][row][k(padded)]
  __shared__ float sPartial[8][16];
  __shared__ float sPos[4][16];

  const int tid   = threadIdx.x;
  const int wave  = tid >> 5;
  const int lane  = tid & 31;
  const int lhalf = lane & 15;
  const bool hiLn = lane >= 16;
  const int rt    = wave >> 1;                 // row tile within block (0..3)
  const int par   = wave & 1;                  // column-tile parity
  const int r_w   = blockIdx.x * 4 + rt;       // global row tile (0..511)

  // --- A fragment: rows 16*r_w .. +15, all K=256, resident whole kernel.
  // (ISA 7.12.2, 16-bit A 16x32): lane<16 -> K {0..7,16..23}+32q,
  // lane>=16 -> K {8..15,24..31}+32q, row M = lane&15.
  v16h afrag[8];
  {
    const _Float16* arow = zh + (size_t)(r_w * 16 + lhalf) * NT_D;
    const int off = hiLn ? 8 : 0;
#pragma unroll
    for (int q = 0; q < 8; ++q) {
      v8h lo = *(const v8h*)(arow + q * 32 + off);
      v8h hi = *(const v8h*)(arow + q * 32 + 16 + off);
      afrag[q] = __builtin_shufflevector(lo, hi,
                   0,1,2,3,4,5,6,7,8,9,10,11,12,13,14,15);
    }
  }

  const int cpos = (r_w + NT_BBLK) & (NT_NBLK - 1);
  // C/D layout: lane<16 -> (M=v, N=lane); lane>=16 -> (M=v+8, N=lane-16).
  const int diagv = (lane < 8) ? lane : ((lane >= 24) ? (lane - 24) : -1);

  // Async-copy addressing: thread t moves 64 B of global row g = t>>3,
  // segment (t&7)*64 into the padded LDS row (conflict-free layout).
  const uint32_t ldsRowOff =
      (uint32_t)(uintptr_t)((const char*)&sB[0][0][0][0]) +
      (uint32_t)((tid >> 3) * LDS_ROWB + (tid & 7) * 64);
  const char* gbase = (const char*)zh + (size_t)(tid >> 3) * (NT_D * 2)
                                      + (size_t)(tid & 7) * 64;

#define NT_ISSUE(cc, b)                                                       \
  do {                                                                        \
    const char* g = gbase + (size_t)(cc) * (32 * NT_D * 2);                   \
    uint32_t l = ldsRowOff + (uint32_t)(b) * (uint32_t)(2 * BUF_HALF * 2);    \
    async_b128(l,      g);                                                    \
    async_b128(l + 16, g + 16);                                               \
    async_b128(l + 32, g + 32);                                               \
    async_b128(l + 48, g + 48);                                               \
  } while (0)

  float s[8];
#pragma unroll
  for (int v = 0; v < 8; ++v) s[v] = 0.0f;

  NT_ISSUE(0, 0);                               // prologue: tile-pair 0 -> buf0

  for (int cc = 0; cc < NT_STEPS; ++cc) {
    const int cur = cc & 1;
    if (cc < NT_STEPS - 1) {                    // prefetch next pair
      NT_ISSUE(cc + 1, cur ^ 1);
      wait_asynccnt<4>();                       // only newest 4 outstanding
    } else {
      wait_asynccnt<0>();
    }
    __syncthreads();                            // tile-pair `cc` visible to all

    // ---- stage ALL 8 B fragments; fence so the scheduler cannot sink the
    // ds_loads back between the WMMAs (one ds clause + one dscnt wait) ----
    const _Float16* brow = &sB[cur][par][lhalf][0] + (hiLn ? 8 : 0);
    v16h bfrag[8];
#pragma unroll
    for (int q = 0; q < 8; ++q) {
      v8h lo = *(const v8h*)(brow + q * 32);
      v8h hi = *(const v8h*)(brow + q * 32 + 16);
      bfrag[q] = __builtin_shufflevector(lo, hi,
                   0,1,2,3,4,5,6,7,8,9,10,11,12,13,14,15);
    }
    sched_fence();

    // ---- 8 back-to-back WMMAs (C-accumulate chain only) ----
    v8f acc = {};
#pragma unroll
    for (int q = 0; q < 8; ++q)
      acc = __builtin_amdgcn_wmma_f32_16x16x32_f16(
              false, afrag[q], false, bfrag[q], (short)0, acc, false, false);
    sched_fence();

    const int c = 2 * cc + par;
    const bool selfTile = (c == r_w);
    const bool posTile  = (c == cpos);
#pragma unroll
    for (int v = 0; v < 8; ++v) {
      const float x = acc[v] * NT_INVT;
      const bool isdiag = (v == diagv);
      if (posTile && isdiag) sPos[rt][hiLn ? (lane - 16) : lane] = x;
      if (!(selfTile && isdiag)) s[v] += __expf(x - NT_SHIFT);
    }
    __syncthreads();                            // done reading buf before reuse
  }
#undef NT_ISSUE

  // Reduce per-lane partials across each 16-lane half (rows 0-7 in lanes
  // 0-15, rows 8-15 in lanes 16-31), stage per-wave sums in LDS.
#pragma unroll
  for (int m = 1; m < 16; m <<= 1)
#pragma unroll
    for (int v = 0; v < 8; ++v) s[v] += __shfl_xor(s[v], m, 32);
  if (lhalf == 0) {
#pragma unroll
    for (int v = 0; v < 8; ++v) sPartial[wave][(hiLn ? 8 : 0) + v] = s[v];
  }
  __syncthreads();

  // Combine even/odd parity waves of each row tile; emit per-row loss.
  if (tid < 64) {
    const int rtt = tid >> 4, m = tid & 15;
    const float t = sPartial[2 * rtt][m] + sPartial[2 * rtt + 1][m];
    const float lse = NT_SHIFT + __logf(t);
    row_loss[blockIdx.x * 64 + tid] = lse - sPos[rtt][m];
  }
}

// ---------------- Kernel 3: deterministic mean reduction --------------------
__global__ void nt_reduce(const float* __restrict__ row_loss,
                          float* __restrict__ out) {
  __shared__ float sm[256];
  float acc = 0.0f;
  for (int i = threadIdx.x; i < NT_N; i += 256) acc += row_loss[i];
  sm[threadIdx.x] = acc;
  __syncthreads();
  for (int s = 128; s > 0; s >>= 1) {
    if ((int)threadIdx.x < s) sm[threadIdx.x] += sm[threadIdx.x + s];
    __syncthreads();
  }
  if (threadIdx.x == 0) out[0] = sm[0] / (float)NT_N;
}

// ---------------------------------------------------------------------------
extern "C" void kernel_launch(void* const* d_in, const int* in_sizes, int n_in,
                              void* d_out, int out_size, void* d_ws, size_t ws_size,
                              hipStream_t stream) {
  (void)in_sizes; (void)n_in; (void)out_size; (void)ws_size;
  const float* z1 = (const float*)d_in[0];
  const float* z2 = (const float*)d_in[1];
  float* out = (float*)d_out;

  _Float16* zh       = (_Float16*)d_ws;                                  // 4 MB
  float*    row_loss = (float*)((char*)d_ws + (size_t)NT_N * NT_D * 2);  // 32 KB

  nt_normalize<<<NT_N / 8, 256, 0, stream>>>(z1, z2, zh);
  nt_lse<<<NT_NBLK / 4, 256, 0, stream>>>(zh, row_loss);
  nt_reduce<<<1, 256, 0, stream>>>(row_loss, out);
}
